// InterestEvolutionLayer_88648124989842
// MI455X (gfx1250) — matrix operand order repaired and checked by reading, not compiled, not run on hardware
//
#include <hip/hip_runtime.h>

// ---------------------------------------------------------------------------
// CDNA5 (gfx1250): f16 WMMA (f32 accum) GEMMs + Tensor Data Mover (TDM)
// async global->LDS staging with TENSORcnt double buffering.
// Shapes: B=4096, T=200, D=U=128 (derived from in_sizes at launch).
// ---------------------------------------------------------------------------

typedef _Float16 h16 __attribute__((ext_vector_type(16)));
typedef _Float16 h8  __attribute__((ext_vector_type(8)));
typedef float    f8  __attribute__((ext_vector_type(8)));
typedef unsigned int u32;
typedef unsigned long long u64;
typedef u32 u32x4 __attribute__((ext_vector_type(4)));
typedef u32 u32x8 __attribute__((ext_vector_type(8)));

__device__ __forceinline__ int lane_id() { return (int)(threadIdx.x & 31); }

__device__ __forceinline__ f8 wmma_f16(h16 a, h16 b, f8 c) {
  return __builtin_amdgcn_wmma_f32_16x16x32_f16(false, a, false, b,
                                                (short)0, c, false, false);
}

// Wave-local LDS producer->consumer fence (CDNA5 split counters).
__device__ __forceinline__ void lds_wave_fence() {
  asm volatile("s_wait_dscnt 0" ::: "memory");
}

__device__ __forceinline__ u32 lds_addr_of(const void* p) {
  // addrspace(3) offset == low 32 bits of the generic address (ISA 10.2).
  return (u32)(u64)(uintptr_t)p;
}

// ---------------------------------------------------------------------------
// TDM: issue a 2D tile load global -> LDS (async, TENSORcnt-tracked).
// D# packing per CDNA5 ISA.
//   data_size_code: 0=1B 1=2B 2=4B 3=8B
//   tile_w elements per row (tile_dim0, 16-bit), tile_h rows (tile_dim1),
//   row_stride in elements (tensor_dim0_stride, 48-bit).
// tensor dims are set == tile dims (tile fully in-bounds, no OOB zeroing).
// ---------------------------------------------------------------------------
__device__ __forceinline__ void tdm_load_2d(u32 lds_addr, u64 gaddr,
                                            u32 data_size_code,
                                            u32 tile_w, u32 tile_h,
                                            u64 row_stride_elems) {
  u32x4 g0;
  g0[0] = 1u;                                        // count=1, user mode
  g0[1] = lds_addr;                                  // lds_addr (bytes)
  g0[2] = (u32)(gaddr & 0xffffffffu);                // global_addr[31:0]
  g0[3] = (u32)((gaddr >> 32) & 0x1ffffffu) | (2u << 30);  // [56:32], type=2
  u32x8 g1;
  g1[0] = (data_size_code & 3u) << 16;               // wg_mask=0, data_size
  g1[1] = (tile_w & 0xffffu) << 16;                  // tensor_dim0[15:0]
  g1[2] = ((tile_w >> 16) & 0xffffu) | ((tile_h & 0xffffu) << 16);
  g1[3] = ((tile_h >> 16) & 0xffffu) | ((tile_w & 0xffffu) << 16); // tile_dim0
  g1[4] = (tile_h & 0xffffu);                        // tile_dim1, tile_dim2=0
  g1[5] = (u32)(row_stride_elems & 0xffffffffu);     // dim0_stride[31:0]
  g1[6] = (u32)((row_stride_elems >> 32) & 0xffffu); // dim0_stride[47:32]
  g1[7] = 0u;                                        // dim1_stride=0
  asm volatile("tensor_load_to_lds %0, %1" :: "s"(g0), "s"(g1) : "memory");
}

// A fragment (16x32 f16) from row-major f16 LDS tile [16][ld], k-chunk kc.
__device__ __forceinline__ h16 lds_load_A(const _Float16* tile, int ld, int kc) {
  int l = lane_id();
  int m  = l & 15;
  int kb = (l < 16) ? 0 : 8;
  const _Float16* p = tile + m * ld + kc * 32 + kb;
  h8 lo = *(const h8*)p;
  h8 hi = *(const h8*)(p + 16);
  h16 a;
#pragma unroll
  for (int j = 0; j < 8; ++j) { a[j] = lo[j]; a[j + 8] = hi[j]; }
  return a;
}

// Same, converting from an f32 LDS tile.
__device__ __forceinline__ h16 lds_load_A_f32(const float* tile, int ld, int kc) {
  int l = lane_id();
  int m  = l & 15;
  int kb = (l < 16) ? 0 : 8;
  const float* p = tile + m * ld + kc * 32 + kb;
  f8 lo = *(const f8*)p;
  f8 hi = *(const f8*)(p + 16);
  h16 a;
#pragma unroll
  for (int j = 0; j < 8; ++j) { a[j] = (_Float16)lo[j]; a[j + 8] = (_Float16)hi[j]; }
  return a;
}

// B fragment (32x16 f16) from N-major (transposed) f16 weights wt[n][k].
__device__ __forceinline__ h16 lds_load_B(const _Float16* wt, int ldk, int nt, int kc) {
  int l = lane_id();
  int n  = l & 15;
  int kb = (l < 16) ? 0 : 8;
  const _Float16* p = wt + (size_t)(nt * 16 + n) * ldk + kc * 32 + kb;
  h8 lo = *(const h8*)p;
  h8 hi = *(const h8*)(p + 16);
  h16 b;
#pragma unroll
  for (int j = 0; j < 8; ++j) { b[j] = lo[j]; b[j + 8] = hi[j]; }
  return b;
}

__device__ __forceinline__ float sigmoidf_(float x) {
  return 1.0f / (1.0f + __expf(-x));
}
__device__ __forceinline__ float dicef_(float x, float alpha) {
  float p = sigmoidf_(x * 0.9999999995f);
  return alpha * (1.0f - p) * x + p * x;
}

// ---------------------------------------------------------------------------
// Kernel 1: pack weights fp32 -> f16 transposed (N-major) into workspace.
// ---------------------------------------------------------------------------
__global__ void prep_weights_kernel(const float* __restrict__ W1,
                                    const float* __restrict__ W2,
                                    const float* __restrict__ Wu,
                                    const float* __restrict__ Wr,
                                    const float* __restrict__ Wc,
                                    _Float16* __restrict__ Wt1h,
                                    _Float16* __restrict__ Wt2h,
                                    _Float16* __restrict__ WtGh) {
  int i = blockIdx.x * blockDim.x + threadIdx.x;
  const int N1 = 512 * 64;
  const int N2 = 16 * 64;
  const int NG = 3 * 128 * 256;
  if (i < N1) {
    int n = i / 512, k = i % 512;
    Wt1h[i] = (_Float16)W1[k * 64 + n];
  } else if (i < N1 + N2) {
    int j = i - N1;
    int n = j / 64, k = j % 64;
    Wt2h[j] = (_Float16)W2[k * 16 + n];
  } else if (i < N1 + N2 + NG) {
    int g = i - N1 - N2;
    int gate = g / (128 * 256);
    int rem  = g % (128 * 256);
    int n = rem / 256, k = rem % 256;
    const float* W = (gate == 0) ? Wu : ((gate == 1) ? Wr : Wc);
    WtGh[g] = (_Float16)W[k * 128 + n];
  }
}

// ---------------------------------------------------------------------------
// Kernel 2: attention scores. One 16-row tile (row = b*T+t) per wave.
// k-tile (8KB contiguous) arrives via TDM, double-buffered over the tile loop.
// ---------------------------------------------------------------------------
__global__ __launch_bounds__(256) void attn_kernel(
    const float* __restrict__ seq,     // [B,T,128]
    const float* __restrict__ tgt,     // [B,128]
    const int*   __restrict__ slen,    // [B]
    const _Float16* __restrict__ Wt1h, // [64][512] N-major f16 (ws)
    const float* __restrict__ alpha1, const float* __restrict__ alpha2,
    const float* __restrict__ W3,
    float* __restrict__ att,           // [B*T]
    int B, int T, int ntiles) {
  extern __shared__ char smem[];
  _Float16* sW1 = (_Float16*)smem;          // 64*512 halves (64 KB)
  _Float16* sW2 = sW1 + 64 * 512;           // 16*64 halves (2 KB), contiguous
  const int PER_WAVE = 8192 + 8192 + 8192 + 2048;  // bytes
  int wid = (int)(threadIdx.x >> 5);
  char* wb = smem + 65536 + 2048 + (size_t)wid * PER_WAVE;
  float* qf32 = (float*)wb;                 // 16x128 f32 q tile
  float* kf0  = qf32 + 2048;                // k tile buffer 0 (f32)
  float* kf1  = kf0 + 2048;                 // k tile buffer 1 (f32)
  _Float16* sh1 = (_Float16*)(kf1 + 2048);  // 16x64 f16 h1 staging

  // Weights: one TDM DMA (66,560 B = 8320 x 8B elements), wave 0 issues.
  if (wid == 0) {
    tdm_load_2d(lds_addr_of(sW1), (u64)(uintptr_t)Wt1h, 3u, 8320u, 1u, 8320u);
    __builtin_amdgcn_s_wait_tensorcnt(0);
  }
  __syncthreads();

  const int l  = lane_id();
  const int n  = l & 15;
  const int mb = (l < 16) ? 0 : 8;
  float a1v[4];
#pragma unroll
  for (int nt = 0; nt < 4; ++nt) a1v[nt] = alpha1[nt * 16 + n];
  const float a2v = alpha2[n];
  const float w3v = W3[n];

  const int stride = (int)((gridDim.x * blockDim.x) >> 5);
  int tile = (int)(blockIdx.x * (blockDim.x >> 5)) + wid;
  int pb = 0;
  if (tile < ntiles)
    tdm_load_2d(lds_addr_of(kf0), (u64)(uintptr_t)(seq + (size_t)tile * 16 * 128),
                2u, 2048u, 1u, 2048u);

  for (; tile < ntiles; tile += stride) {
    const int row0 = tile * 16;
    float* kf = pb ? kf1 : kf0;
    // Prefetch next tile's k data into the other buffer (TDM async).
    int nxt = tile + stride;
    if (nxt < ntiles) {
      tdm_load_2d(lds_addr_of(pb ? kf0 : kf1),
                  (u64)(uintptr_t)(seq + (size_t)nxt * 16 * 128),
                  2u, 2048u, 1u, 2048u);
      __builtin_amdgcn_s_wait_tensorcnt(1);   // current tile's DMA done
    } else {
      __builtin_amdgcn_s_wait_tensorcnt(0);
    }

    // Stage q tile (coalesced float4 loads).
    for (int idx = l; idx < 512; idx += 32) {
      int m = idx >> 5;
      int d4 = (idx & 31) * 4;
      int b = (row0 + m) / T;
      *(float4*)(qf32 + m * 128 + d4) = *(const float4*)(tgt + (size_t)b * 128 + d4);
    }
    lds_wave_fence();

    // GEMM1: [16,512] x [512,64]; chunk kc covers features f = 32*kc..32*kc+31
    f8 acc[4] = {f8{}, f8{}, f8{}, f8{}};
#pragma unroll
    for (int kc = 0; kc < 16; ++kc) {
      const int region = kc >> 2;            // 0:q 1:k 2:q-k 3:q*k
      const int dbase  = (kc & 3) * 32;
      const int m  = l & 15;
      const int kb = (l < 16) ? 0 : 8;
      const float* pq = qf32 + m * 128 + dbase + kb;
      const float* pk = kf   + m * 128 + dbase + kb;
      f8 ql = *(const f8*)pq, qh = *(const f8*)(pq + 16);
      f8 kl = *(const f8*)pk, kh = *(const f8*)(pk + 16);
      h16 a;
#pragma unroll
      for (int j = 0; j < 8; ++j) {
        float v0, v1;
        if (region == 0)      { v0 = ql[j];         v1 = qh[j]; }
        else if (region == 1) { v0 = kl[j];         v1 = kh[j]; }
        else if (region == 2) { v0 = ql[j] - kl[j]; v1 = qh[j] - kh[j]; }
        else                  { v0 = ql[j] * kl[j]; v1 = qh[j] * kh[j]; }
        a[j] = (_Float16)v0; a[j + 8] = (_Float16)v1;
      }
#pragma unroll
      for (int nt = 0; nt < 4; ++nt)
        acc[nt] = wmma_f16(a, lds_load_B(sW1, 512, nt, kc), acc[nt]);
    }

    // Dice + stage h1 (f16) in A layout for GEMM2.
#pragma unroll
    for (int nt = 0; nt < 4; ++nt) {
#pragma unroll
      for (int i = 0; i < 8; ++i) {
        float y = dicef_(acc[nt][i], a1v[nt]);
        sh1[(mb + i) * 64 + nt * 16 + n] = (_Float16)y;
      }
    }
    lds_wave_fence();

    // GEMM2: [16,64] x [64,16]
    f8 acc2 = {};
#pragma unroll
    for (int kc = 0; kc < 2; ++kc)
      acc2 = wmma_f16(lds_load_A(sh1, 64, kc), lds_load_B(sW2, 64, 0, kc), acc2);

    // Dice + GEMM3 (dot W3, width-16 shuffle reduce) + mask + sigmoid.
#pragma unroll
    for (int i = 0; i < 8; ++i) {
      float y = dicef_(acc2[i], a2v) * w3v;
      y += __shfl_xor(y, 1, 16);
      y += __shfl_xor(y, 2, 16);
      y += __shfl_xor(y, 4, 16);
      y += __shfl_xor(y, 8, 16);
      if (n == 0) {
        int row = row0 + mb + i;
        int b = row / T;
        int t = row - b * T;
        att[row] = (t < slen[b]) ? sigmoidf_(y) : 0.0f;  // sigmoid(NEG_BIG)==0
      }
    }
    pb ^= 1;
  }
}

// ---------------------------------------------------------------------------
// Kernel 3: AUGRU scan. 4 waves/block; each wave scans 16 batch rows over T.
// x_t tiles arrive via 2D TDM (16 rows x 128 f32, row stride T*128 elems),
// double-buffered across timesteps. h in LDS f32; u accums kept in VGPRs.
// ---------------------------------------------------------------------------
__global__ __launch_bounds__(128) void augru_kernel(
    const float* __restrict__ seq,     // [B,T,128]
    const float* __restrict__ att,     // [B*T]
    const int*   __restrict__ slen,    // [B]
    const _Float16* __restrict__ WtGh, // [3][128][256] N-major f16 (ws)
    float* __restrict__ out,           // [B,128]
    int B, int T) {
  extern __shared__ char smem[];
  _Float16* sW = (_Float16*)smem;           // 3*128*256 halves (192 KB)
  const int PER_WAVE = 8192 + 8192 + 4096 + 8192;  // bytes
  int wid = (int)(threadIdx.x >> 5);
  char* wb = smem + (size_t)3 * 128 * 256 * 2 + (size_t)wid * PER_WAVE;
  float* sxf0 = (float*)wb;                 // x tile buffer 0 (16x128 f32)
  float* sxf1 = sxf0 + 2048;                // x tile buffer 1
  _Float16* srh = (_Float16*)(sxf1 + 2048); // 16x128 f16 r*h
  float* sh = (float*)(srh + 2048);         // 16x128 f32 hidden state

  // Weights: one TDM DMA (196,608 B as 6 x 4096 x 8B), wave 0 issues.
  if (wid == 0) {
    tdm_load_2d(lds_addr_of(sW), (u64)(uintptr_t)WtGh, 3u, 4096u, 6u, 4096u);
    __builtin_amdgcn_s_wait_tensorcnt(0);
  }

  const int l  = lane_id();
  const int n  = l & 15;
  const int mb = (l < 16) ? 0 : 8;
  const int rowBase = (int)blockIdx.x * ((int)blockDim.x >> 5) * 16 + wid * 16;

  for (int idx = l; idx < 2048; idx += 32) sh[idx] = 0.0f;
  int slv[8];
#pragma unroll
  for (int i = 0; i < 8; ++i) slv[i] = slen[rowBase + mb + i];
  __syncthreads();

  const _Float16* Wu = sW;
  const _Float16* Wr = sW + 128 * 256;
  const _Float16* Wc = sW + 2 * 128 * 256;

  // Prologue: DMA x_0 into buffer 0.
  tdm_load_2d(lds_addr_of(sxf0),
              (u64)(uintptr_t)(seq + (size_t)rowBase * T * 128),
              2u, 128u, 16u, (u64)T * 128u);

  for (int t = 0; t < T; ++t) {
    float* sx = (t & 1) ? sxf1 : sxf0;
    // Async prefetch of x_{t+1} into the other buffer while we compute.
    if (t + 1 < T) {
      tdm_load_2d(lds_addr_of((t & 1) ? sxf0 : sxf1),
                  (u64)(uintptr_t)(seq + ((size_t)rowBase * T + (t + 1)) * 128),
                  2u, 128u, 16u, (u64)T * 128u);
      __builtin_amdgcn_s_wait_tensorcnt(1);   // x_t ready
    } else {
      __builtin_amdgcn_s_wait_tensorcnt(0);
    }

    float av[8];
#pragma unroll
    for (int i = 0; i < 8; ++i) av[i] = att[(size_t)(rowBase + mb + i) * T + t];

    // A fragments for gi = [x | h] (K=256, 8 chunks).
    h16 agi[8];
#pragma unroll
    for (int kc = 0; kc < 4; ++kc) agi[kc] = lds_load_A_f32(sx, 128, kc);
#pragma unroll
    for (int kc = 0; kc < 4; ++kc) agi[4 + kc] = lds_load_A_f32(sh, 128, kc);

    // u and r gates: u accumulators stay in VGPRs; r*h staged to LDS (f16).
    f8 uacc[8];
#pragma unroll
    for (int nt = 0; nt < 8; ++nt) {
      f8 au = {}, ar = {};
#pragma unroll
      for (int kc = 0; kc < 8; ++kc) {
        au = wmma_f16(agi[kc], lds_load_B(Wu, 256, nt, kc), au);
        ar = wmma_f16(agi[kc], lds_load_B(Wr, 256, nt, kc), ar);
      }
      uacc[nt] = au;
#pragma unroll
      for (int i = 0; i < 8; ++i) {
        int m = mb + i;
        int col = nt * 16 + n;
        srh[m * 128 + col] = (_Float16)(sigmoidf_(ar[i]) * sh[m * 128 + col]);
      }
    }
    lds_wave_fence();

    // Candidate: gi2 = [x | r*h]; x-part fragments reused.
#pragma unroll
    for (int kc = 0; kc < 4; ++kc) agi[4 + kc] = lds_load_A(srh, 128, kc);
#pragma unroll
    for (int nt = 0; nt < 8; ++nt) {
      f8 ac = {};
#pragma unroll
      for (int kc = 0; kc < 8; ++kc)
        ac = wmma_f16(agi[kc], lds_load_B(Wc, 256, nt, kc), ac);
#pragma unroll
      for (int i = 0; i < 8; ++i) {
        int m = mb + i;
        int col = nt * 16 + n;
        float valid = (t < slv[i]) ? 1.0f : 0.0f;
        float uef = sigmoidf_(uacc[nt][i]) * av[i] * valid;  // masked AUGRU
        float c = tanhf(ac[i]);
        float hold = sh[m * 128 + col];
        sh[m * 128 + col] = hold + uef * (c - hold);
      }
    }
    lds_wave_fence();   // drain LDS before next TDM overwrites old x buffer
  }

  // Emit final hidden state.
  for (int idx = l; idx < 2048; idx += 32) {
    int m = idx >> 7, d = idx & 127;
    out[(size_t)(rowBase + m) * 128 + d] = sh[idx];
  }
}

// ---------------------------------------------------------------------------
extern "C" void kernel_launch(void* const* d_in, const int* in_sizes, int n_in,
                              void* d_out, int out_size, void* d_ws, size_t ws_size,
                              hipStream_t stream) {
  const float* seq   = (const float*)d_in[0];   // [B,T,128]
  const float* tgt   = (const float*)d_in[1];   // [B,128]
  const int*   slen  = (const int*)d_in[2];     // [B,1]
  const float* W1    = (const float*)d_in[3];   // [512,64]
  const float* a1    = (const float*)d_in[4];   // [64]
  const float* W2    = (const float*)d_in[5];   // [64,16]
  const float* a2    = (const float*)d_in[6];   // [16]
  const float* W3    = (const float*)d_in[7];   // [16,1]
  const float* Wu    = (const float*)d_in[8];   // [256,128]
  const float* Wr    = (const float*)d_in[9];   // [256,128]
  const float* Wc    = (const float*)d_in[10];  // [256,128]

  const int D = 128;
  const int B = in_sizes[1] / D;
  const int T = in_sizes[0] / (B * D);

  // Workspace layout (Wt1h and Wt2h deliberately contiguous for one TDM DMA).
  char* ws = (char*)d_ws;
  float* att = (float*)ws;
  size_t off = ((size_t)B * T * sizeof(float) + 255) & ~(size_t)255;
  _Float16* Wt1h = (_Float16*)(ws + off); off += (size_t)512 * 64 * 2;
  _Float16* Wt2h = (_Float16*)(ws + off); off += (size_t)16 * 64 * 2;
  _Float16* WtGh = (_Float16*)(ws + off); off += (size_t)3 * 128 * 256 * 2;
  (void)ws_size;

  // 1) Weight prep.
  {
    const int total = 512 * 64 + 16 * 64 + 3 * 128 * 256;
    prep_weights_kernel<<<(total + 255) / 256, 256, 0, stream>>>(
        W1, W2, Wu, Wr, Wc, Wt1h, Wt2h, WtGh);
  }

  // 2) Attention scores.
  {
    const int ntiles = (B * T) / 16;
    const size_t smem = 65536 + 2048 + (size_t)8 * (8192 + 8192 + 8192 + 2048);
    attn_kernel<<<640, 256, smem, stream>>>(seq, tgt, slen, Wt1h,
                                            a1, a2, W3, att, B, T, ntiles);
  }

  // 3) AUGRU scan -> d_out.
  {
    const int blocks = B / 64;   // 4 waves x 16 rows per block
    const size_t smem = (size_t)3 * 128 * 256 * 2 +
                        (size_t)4 * (8192 + 8192 + 4096 + 8192);
    augru_kernel<<<blocks, 128, smem, stream>>>(seq, att, slen, WtGh,
                                                (float*)d_out, B, T);
  }
  (void)n_in; (void)out_size;
}
